// LSTMAttentionDecoder_33784212750988
// MI455X (gfx1250) — compile-verified, matrix operand order checked by reading, not run on hardware
//
#include <hip/hip_runtime.h>
#include <math.h>

// ---------------- types ----------------
typedef __bf16 bf16t;
typedef bf16t v16bf __attribute__((ext_vector_type(16)));
typedef float v8f   __attribute__((ext_vector_type(8)));
typedef unsigned short u16;
typedef u16 v16u    __attribute__((ext_vector_type(16)));

#define Bm 32
#define Tm 128
#define Sm 256
#define Dm 1024
#define Vm 10000
#define Lm 2

__device__ __forceinline__ u16 f2bf(float f) {
    unsigned x = __builtin_bit_cast(unsigned, f);
    unsigned r = (x + 0x7FFFu + ((x >> 16) & 1u)) >> 16;   // round-to-nearest-even
    return (u16)r;
}

#define WMMA_BF16(a, b, c) \
    __builtin_amdgcn_wmma_f32_16x16x32_bf16(false, (a), false, (b), (short)0, (c), false, false)

// ---------------- fp32 -> bf16 bulk conversion ----------------
__global__ __launch_bounds__(256) void k_f32_to_bf16(const float* __restrict__ in,
                                                     u16* __restrict__ out, long n) {
    long i = (long)blockIdx.x * blockDim.x + threadIdx.x;
    long stride = (long)gridDim.x * blockDim.x;
    for (; i < n; i += stride) out[i] = f2bf(in[i]);
}

// ---------------- fill zeros ----------------
__global__ __launch_bounds__(256) void k_fill0(float* __restrict__ p, long n) {
    long i = (long)blockIdx.x * blockDim.x + threadIdx.x;
    long stride = (long)gridDim.x * blockDim.x;
    for (; i < n; i += stride) p[i] = 0.f;
}

// ---------------- embedding gather (emit bf16) ----------------
__global__ __launch_bounds__(256) void k_embed(const int* __restrict__ tgt,
                                               const float* __restrict__ emb,
                                               u16* __restrict__ xall) {
    int bt = blockIdx.x;                       // b*T + t
    int row = tgt[bt];
    const float* src = emb + (size_t)row * Dm;
    u16* dst = xall + (size_t)bt * Dm;
    for (int d = threadIdx.x; d < Dm; d += 256) dst[d] = f2bf(src[d]);
}

// ---------------- one K-pass of the 2x4-tile GEMM ----------------
// Fragment scheme (per-lane contiguous): lanes 0-15 -> row r, K..K+15 ;
// lanes 16-31 -> row r, K+16..K+31.  B comes straight from row-major (N,K) weights.
__device__ __forceinline__ void gemm_pass(
    const u16* __restrict__ A, int lda, const u16* __restrict__ W, int ldw,
    int tm0, const int* tnc, int r, int ko, int K,
    v8f& c00, v8f& c01, v8f& c02, v8f& c03,
    v8f& c10, v8f& c11, v8f& c12, v8f& c13) {
    const u16* a0 = A + (size_t)(tm0 * 16 + r) * lda + ko;
    const u16* a1 = A + (size_t)(tm0 * 16 + 16 + r) * lda + ko;
    const u16* w0 = W + (size_t)(tnc[0] * 16 + r) * ldw + ko;
    const u16* w1 = W + (size_t)(tnc[1] * 16 + r) * ldw + ko;
    const u16* w2 = W + (size_t)(tnc[2] * 16 + r) * ldw + ko;
    const u16* w3 = W + (size_t)(tnc[3] * 16 + r) * ldw + ko;
    for (int kk = 0; kk < K; kk += 32) {
        v16bf fa0 = __builtin_bit_cast(v16bf, *(const v16u*)(a0 + kk));
        v16bf fa1 = __builtin_bit_cast(v16bf, *(const v16u*)(a1 + kk));
        v16bf fb0 = __builtin_bit_cast(v16bf, *(const v16u*)(w0 + kk));
        v16bf fb1 = __builtin_bit_cast(v16bf, *(const v16u*)(w1 + kk));
        v16bf fb2 = __builtin_bit_cast(v16bf, *(const v16u*)(w2 + kk));
        v16bf fb3 = __builtin_bit_cast(v16bf, *(const v16u*)(w3 + kk));
        c00 = WMMA_BF16(fa0, fb0, c00);
        c10 = WMMA_BF16(fa1, fb0, c10);
        c01 = WMMA_BF16(fa0, fb1, c01);
        c11 = WMMA_BF16(fa1, fb1, c11);
        c02 = WMMA_BF16(fa0, fb2, c02);
        c12 = WMMA_BF16(fa1, fb2, c12);
        c03 = WMMA_BF16(fa0, fb3, c03);
        c13 = WMMA_BF16(fa1, fb3, c13);
    }
}

// ---------------- dual-input bf16 WMMA GEMM, 2x4 tiles per wave ----------------
// C[M,N] = act( A0[M,K0] * W0[N,K0]^T + A1[M,K1] * W1[N,K1]^T + bias[N] )
// A/W bf16 (u16 storage), row-major.  Each wave: two 16-row M tiles x four 16-col
// N tiles (B fragments reused across M).  4 waves/block.  Requires M % 32 == 0.
__global__ __launch_bounds__(128)
void k_gemm_bf16(const u16* __restrict__ A0, const u16* __restrict__ W0,
                 const u16* __restrict__ A1, const u16* __restrict__ W1,
                 const float* __restrict__ bias,
                 float* __restrict__ C, u16* __restrict__ Cbf,
                 int M, int N, int K0, int K1,
                 int lda0, int lda1, int ldw0, int ldw1, int ldc, int act) {
    const int lane = threadIdx.x & 31;
    const int wave = threadIdx.x >> 5;
    const int ntn  = N >> 4;
    const int tn0  = (blockIdx.x * 4 + wave) * 4;        // first of 4 N tiles
    if (tn0 >= ntn) return;
    const int tm0 = blockIdx.y * 2;                      // two M tiles per wave

    const int r  = lane & 15;          // row within 16-row tile
    const int ko = (lane >> 4) << 4;   // lanes 16-31 take the K+16 half

    // clamp invalid tail tiles onto a valid one (stores are guarded)
    int tnc[4];
#pragma unroll
    for (int j = 0; j < 4; ++j) {
        int tn = tn0 + j;
        tnc[j] = tn < ntn ? tn : ntn - 1;
    }

    v8f c00 = {}, c01 = {}, c02 = {}, c03 = {};
    v8f c10 = {}, c11 = {}, c12 = {}, c13 = {};

    gemm_pass(A0, lda0, W0, ldw0, tm0, tnc, r, ko, K0,
              c00, c01, c02, c03, c10, c11, c12, c13);
    if (A1)
        gemm_pass(A1, lda1, W1, ldw1, tm0, tnc, r, ko, K1,
                  c00, c01, c02, c03, c10, c11, c12, c13);

    // C layout: VGPR i, lanes 0-15 -> M=i,N=lane ; lanes 16-31 -> M=8+i,N=lane-16
    const int mh = (lane >> 4) << 3;
    v8f* accs[8] = { &c00, &c01, &c02, &c03, &c10, &c11, &c12, &c13 };
#pragma unroll
    for (int mt = 0; mt < 2; ++mt) {
        const int mb = (tm0 + mt) * 16 + mh;
#pragma unroll
        for (int j = 0; j < 4; ++j) {
            int tn = tn0 + j;
            if (tn >= ntn) break;
            int n = tn * 16 + r;
            float bv = bias ? bias[n] : 0.f;
            v8f& a = *accs[mt * 4 + j];
#pragma unroll
            for (int i = 0; i < 8; ++i) {
                float v = a[i] + bv;
                if (act == 1) v = tanhf(v);
                if (C)   C[(size_t)(mb + i) * ldc + n] = v;
                if (Cbf) Cbf[(size_t)(mb + i) * ldc + n] = f2bf(v);
            }
        }
    }
}

// ---------------- LSTM cell pointwise (writes bf16 h, f32 c) ----------------
__global__ __launch_bounds__(256)
void k_lstm_cell(const float* __restrict__ gates, const float* __restrict__ b_hh,
                 u16* __restrict__ h_bf, float* __restrict__ c) {
    int idx = blockIdx.x * 256 + threadIdx.x;     // over B*D = 32768
    int b = idx >> 10, d = idx & 1023;
    const float* g = gates + (size_t)b * (4 * Dm);
    float gi = g[d]            + b_hh[d];
    float gf = g[Dm + d]       + b_hh[Dm + d];
    float gg = g[2 * Dm + d]   + b_hh[2 * Dm + d];
    float go = g[3 * Dm + d]   + b_hh[3 * Dm + d];
    float si = 1.f / (1.f + expf(-gi));
    float sf = 1.f / (1.f + expf(-gf));
    float so = 1.f / (1.f + expf(-go));
    float cn = sf * c[idx] + si * tanhf(gg);
    float hn = so * tanhf(cn);
    c[idx] = cn;
    h_bf[idx] = f2bf(hn);
}

// ---------------- attention energy: conv(prev attn) + tanh + dot ----------------
__global__ __launch_bounds__(256)
void k_energy(const float* __restrict__ qp, const float* __restrict__ vproj,
              const float* __restrict__ attn_prev,
              const float* __restrict__ wconv, const float* __restrict__ bconv,
              const float* __restrict__ abias, const float* __restrict__ fw,
              const float* __restrict__ fb, float* __restrict__ energy) {
    int s = blockIdx.x, b = blockIdx.y, tid = threadIdx.x;
    float am1 = (s > 0)      ? attn_prev[b * Sm + s - 1] : 0.f;
    float a0  =                attn_prev[b * Sm + s];
    float ap1 = (s < Sm - 1) ? attn_prev[b * Sm + s + 1] : 0.f;
    const float* vp = vproj + ((size_t)(b * Sm + s)) * Dm;
    const float* q  = qp + (size_t)b * Dm;
    float acc = 0.f;
    for (int d = tid; d < Dm; d += 256) {
        float loc = am1 * wconv[d * 3 + 0] + a0 * wconv[d * 3 + 1] + ap1 * wconv[d * 3 + 2] + bconv[d];
        acc += tanhf(q[d] + vp[d] + loc + abias[d]) * fw[d];
    }
    __shared__ float sh[256];
    sh[tid] = acc; __syncthreads();
    for (int st = 128; st > 0; st >>= 1) { if (tid < st) sh[tid] += sh[tid + st]; __syncthreads(); }
    if (tid == 0) energy[b * Sm + s] = sh[0] + fb[0];
}

// ---------------- softmax over S + context (bf16) = attn . enc ----------------
__global__ __launch_bounds__(256)
void k_softmax_ctx(const float* __restrict__ energy, const float* __restrict__ enc,
                   float* __restrict__ attn_new, u16* __restrict__ context_bf) {
    int b = blockIdx.x, s = threadIdx.x;
    __shared__ float sh[256];
    __shared__ float sa[256];
    float x = energy[b * Sm + s];
    sh[s] = x; __syncthreads();
    for (int st = 128; st > 0; st >>= 1) { if (s < st) sh[s] = fmaxf(sh[s], sh[s + st]); __syncthreads(); }
    float mx = sh[0]; __syncthreads();
    float e = expf(x - mx);
    sh[s] = e; __syncthreads();
    for (int st = 128; st > 0; st >>= 1) { if (s < st) sh[s] += sh[s + st]; __syncthreads(); }
    float a = e / sh[0];
    attn_new[b * Sm + s] = a;
    sa[s] = a; __syncthreads();
    const float* eb = enc + (size_t)b * Sm * Dm;
    for (int d = s; d < Dm; d += 256) {
        float acc = 0.f;
        for (int s2 = 0; s2 < Sm; ++s2) acc += sa[s2] * eb[(size_t)s2 * Dm + d];
        context_bf[(size_t)b * Dm + d] = f2bf(acc);
    }
}

// ---------------- log-softmax over V, write output ----------------
__global__ __launch_bounds__(256)
void k_logsoftmax(const float* __restrict__ logits, float* __restrict__ out, int t) {
    int b = blockIdx.x, tid = threadIdx.x;
    const float* row = logits + (size_t)b * Vm;
    __shared__ float sh[256];
    float mx = -3.4e38f;
    for (int v = tid; v < Vm; v += 256) mx = fmaxf(mx, row[v]);
    sh[tid] = mx; __syncthreads();
    for (int st = 128; st > 0; st >>= 1) { if (tid < st) sh[tid] = fmaxf(sh[tid], sh[tid + st]); __syncthreads(); }
    mx = sh[0]; __syncthreads();
    float se = 0.f;
    for (int v = tid; v < Vm; v += 256) se += expf(row[v] - mx);
    sh[tid] = se; __syncthreads();
    for (int st = 128; st > 0; st >>= 1) { if (tid < st) sh[tid] += sh[tid + st]; __syncthreads(); }
    float lse = logf(sh[0]) + mx;
    float* orow = out + ((size_t)b * Tm + t) * Vm;
    for (int v = tid; v < Vm; v += 256) orow[v] = row[v] - lse;
}

// ---------------- host ----------------
extern "C" void kernel_launch(void* const* d_in, const int* in_sizes, int n_in,
                              void* d_out, int out_size, void* d_ws, size_t ws_size,
                              hipStream_t stream) {
    const float* enc     = (const float*)d_in[0];
    const int*   targets = (const int*)d_in[1];
    const float* emb     = (const float*)d_in[2];
    const float* W_ih    = (const float*)d_in[3];
    const float* W_hh    = (const float*)d_in[4];
    const float* b_ih    = (const float*)d_in[5];
    const float* b_hh    = (const float*)d_in[6];
    const float* Wconv   = (const float*)d_in[7];
    const float* bconv   = (const float*)d_in[8];
    const float* Wq      = (const float*)d_in[9];
    const float* Wv      = (const float*)d_in[10];
    const float* abias   = (const float*)d_in[11];
    const float* fw      = (const float*)d_in[12];
    const float* fb      = (const float*)d_in[13];
    const float* fc1_w   = (const float*)d_in[14];
    const float* fc1_b   = (const float*)d_in[15];
    const float* fc2_w   = (const float*)d_in[16];
    const float* fc2_b   = (const float*)d_in[17];
    float* out = (float*)d_out;
    (void)in_sizes; (void)n_in; (void)out_size; (void)ws_size;

    char* wsp = (char*)d_ws;
    auto alloc = [&](size_t bytes) -> char* {
        char* p = wsp; wsp += (bytes + 255) & ~(size_t)255; return p;
    };

    const size_t GW = (size_t)Lm * 4 * Dm * Dm;            // 8.39M elems per weight set
    u16*   wih_bf  = (u16*)  alloc(GW * 2);
    u16*   whh_bf  = (u16*)  alloc(GW * 2);
    u16*   wq_bf   = (u16*)  alloc((size_t)Dm * Dm * 2);
    u16*   wv_bf   = (u16*)  alloc((size_t)Dm * Dm * 2);
    u16*   fc1_bf  = (u16*)  alloc((size_t)Dm * 2 * Dm * 2);
    u16*   fc2_bf  = (u16*)  alloc((size_t)Vm * Dm * 2);
    u16*   enc_bf  = (u16*)  alloc((size_t)Bm * Sm * Dm * 2);
    u16*   embx_bf = (u16*)  alloc((size_t)Bm * Tm * Dm * 2);
    float* vproj   = (float*) alloc((size_t)Bm * Sm * Dm * 4);
    float* gates   = (float*) alloc((size_t)Bm * 4 * Dm * 4);
    u16*   h_bf    = (u16*)  alloc((size_t)Lm * Bm * Dm * 2);   // zero region start
    float* c       = (float*) alloc((size_t)Lm * Bm * Dm * 4);
    float* attnA   = (float*) alloc((size_t)Bm * Sm * 4);        // zero region end
    float* attnB   = (float*) alloc((size_t)Bm * Sm * 4);
    float* qp      = (float*) alloc((size_t)Bm * Dm * 4);
    float* energy  = (float*) alloc((size_t)Bm * Sm * 4);
    u16*   ctx_bf  = (u16*)  alloc((size_t)Bm * Dm * 2);
    u16*   hid_bf  = (u16*)  alloc((size_t)Bm * Dm * 2);
    float* logits  = (float*) alloc((size_t)Bm * Vm * 4);

    // weight + encoder conversion fp32 -> bf16 (L2-resident afterwards: ~77MB << 192MB)
    k_f32_to_bf16<<<2048, 256, 0, stream>>>(W_ih,  wih_bf, (long)GW);
    k_f32_to_bf16<<<2048, 256, 0, stream>>>(W_hh,  whh_bf, (long)GW);
    k_f32_to_bf16<<<1024, 256, 0, stream>>>(Wq,    wq_bf,  (long)Dm * Dm);
    k_f32_to_bf16<<<1024, 256, 0, stream>>>(Wv,    wv_bf,  (long)Dm * Dm);
    k_f32_to_bf16<<<1024, 256, 0, stream>>>(fc1_w, fc1_bf, (long)Dm * 2 * Dm);
    k_f32_to_bf16<<<2048, 256, 0, stream>>>(fc2_w, fc2_bf, (long)Vm * Dm);
    k_f32_to_bf16<<<2048, 256, 0, stream>>>(enc,   enc_bf, (long)Bm * Sm * Dm);

    // zero h_bf, c, attnA (contiguous region; sizes all multiples of 256B)
    long zbytes = (long)(Lm * Bm * Dm * 2 + Lm * Bm * Dm * 4 + Bm * Sm * 4);
    k_fill0<<<512, 256, 0, stream>>>((float*)h_bf, zbytes / 4);

    // embedding for all steps (bf16)
    k_embed<<<Bm * Tm, 256, 0, stream>>>(targets, emb, embx_bf);

    // vproj = enc(8192x1024) @ Wv^T : M=8192, N=1024, K=1024
    k_gemm_bf16<<<dim3(4, 256), 128, 0, stream>>>(
        enc_bf, wv_bf, nullptr, nullptr, nullptr, vproj, nullptr,
        Bm * Sm, Dm, Dm, 0, Dm, 0, Dm, 0, Dm, 0);

    float* a_prev = attnA;
    float* a_next = attnB;

    for (int t = 0; t < Tm; ++t) {
        // ---- LSTM layer 0: gates = x_t @ Wih0^T + h0 @ Whh0^T + b_ih0 ----
        k_gemm_bf16<<<dim3(16, 1), 128, 0, stream>>>(
            embx_bf + (size_t)t * Dm, wih_bf,
            h_bf, whh_bf,
            b_ih, gates, nullptr,
            Bm, 4 * Dm, Dm, Dm,
            Tm * Dm, Dm, Dm, Dm, 4 * Dm, 0);
        k_lstm_cell<<<128, 256, 0, stream>>>(gates, b_hh, h_bf, c);

        // ---- LSTM layer 1 ----
        k_gemm_bf16<<<dim3(16, 1), 128, 0, stream>>>(
            h_bf, wih_bf + GW / 2,
            h_bf + Bm * Dm, whh_bf + GW / 2,
            b_ih + 4 * Dm, gates, nullptr,
            Bm, 4 * Dm, Dm, Dm,
            Dm, Dm, Dm, Dm, 4 * Dm, 0);
        k_lstm_cell<<<128, 256, 0, stream>>>(gates, b_hh + 4 * Dm,
                                             h_bf + Bm * Dm, c + Bm * Dm);

        // ---- qp = q @ Wq^T ----
        k_gemm_bf16<<<dim3(4, 1), 128, 0, stream>>>(
            h_bf + Bm * Dm, wq_bf, nullptr, nullptr, nullptr, qp, nullptr,
            Bm, Dm, Dm, 0, Dm, 0, Dm, 0, Dm, 0);

        // ---- energy(b,s) = fw . tanh(qp + vproj + conv(prev_attn) + abias) + fb ----
        k_energy<<<dim3(Sm, Bm), 256, 0, stream>>>(qp, vproj, a_prev, Wconv, bconv,
                                                   abias, fw, fb, energy);

        // ---- softmax over S + context (bf16) ----
        k_softmax_ctx<<<Bm, 256, 0, stream>>>(energy, enc, a_next, ctx_bf);

        // ---- hid = tanh([q, context] @ fc1_w^T + fc1_b) -> bf16 ----
        k_gemm_bf16<<<dim3(4, 1), 128, 0, stream>>>(
            h_bf + Bm * Dm, fc1_bf,
            ctx_bf, fc1_bf + Dm,
            fc1_b, nullptr, hid_bf,
            Bm, Dm, Dm, Dm,
            Dm, Dm, 2 * Dm, 2 * Dm, Dm, 1);

        // ---- logits = hid @ fc2_w^T + fc2_b : N=10000 (625 tiles) ----
        k_gemm_bf16<<<dim3(40, 1), 128, 0, stream>>>(
            hid_bf, fc2_bf, nullptr, nullptr, fc2_b, logits, nullptr,
            Bm, Vm, Dm, 0, Dm, 0, Dm, 0, Vm, 0);

        // ---- log_softmax -> out[:, t, :] ----
        k_logsoftmax<<<Bm, 256, 0, stream>>>(logits, out, t);

        float* tmp = a_prev; a_prev = a_next; a_next = tmp;
    }
}
